// SuperResolution_6468220748321
// MI455X (gfx1250) — matrix operand order repaired and checked by reading, not compiled, not run on hardware
//
#include <hip/hip_runtime.h>
#include <hip/hip_bf16.h>

// ---------------------------------------------------------------------------
// SuperResolution forward for MI455X (gfx1250, wave32, WMMA bf16).
// All heavy math via v_wmma_f32_16x16x32_bf16 (fp32 accum).
// Activations: NHWC bf16, channels zero-padded to K-tile multiples.
// HBM-bound workload (~9 GB @ 23.3 TB/s); bf16 halves traffic vs fp32.
// v2: software-pipelined fragments (double-buffer), dual accumulators
//     (2 N-tiles/wave), 2 output rows per conv block (halo reuse).
// ---------------------------------------------------------------------------

typedef __attribute__((ext_vector_type(16))) __bf16 v16bf_t;
typedef __attribute__((ext_vector_type(8)))  float  v8f_t;

union FragU { uint4 u[2]; v16bf_t v; };

static __device__ __forceinline__ unsigned short f2bf(float f) {
  unsigned u = __builtin_bit_cast(unsigned, f);
  u += 0x7fffu + ((u >> 16) & 1u);          // round-to-nearest-even
  return (unsigned short)(u >> 16);
}
static __device__ __forceinline__ float bf2f(unsigned short h) {
  return __builtin_bit_cast(float, ((unsigned)h) << 16);
}

#define WMMA_BF16(A, B, C) \
  __builtin_amdgcn_wmma_f32_16x16x32_bf16(false, (A), false, (B), (short)0, (C), false, false)

// ---------------------------------------------------------------------------
// Implicit-GEMM 3x3 conv (pad=1). Block: 2 rows x 64 px x OP channels.
// waves = 8 M-tiles (2 rows x 4 x-tiles) x ceil(NT/2) N-groups.
// Each wave: 1 A fragment feeds 2 independent accumulators (2 N-tiles).
// Weights packed [9*KC][OP][32] bf16 (B-fragment layout).
// ---------------------------------------------------------------------------
template<int KC, int NT>
__global__ void __launch_bounds__(512)
conv3x3_wmma(const unsigned short* __restrict__ in,   // [B,H,W,KC*32] bf16
             const unsigned short* __restrict__ wpk,  // [9*KC][NT*16][32] bf16
             const float* __restrict__ bias, int bias_n,
             const unsigned short* __restrict__ resid,// [B,H,W,NT*16] or null
             unsigned short* __restrict__ out,        // [B,H,W,NT*16]
             int H, int W, int relu)
{
  constexpr int CP = KC * 32;
  constexpr int OP = NT * 16;
  constexpr bool DUAL = (NT > 1);
  constexpr int STEPS = 9 * KC;
  constexpr int IN_BYTES = 4 * 66 * CP * 2;
  constexpr int W_BYTES = 9 * KC * OP * 64;
  __shared__ char smem[IN_BYTES + W_BYTES];

  const int x0 = blockIdx.x * 64;
  const int y0 = blockIdx.y * 2;
  const int b  = blockIdx.z;
  const int tid = threadIdx.x;
  char* ldsIn = smem;
  char* ldsW  = smem + IN_BYTES;

  // Stage input rows y0-1 .. y0+2, cols x0-1..x0+64 (zero halo).
  {
    constexpr int CHUNKS = CP / 4;            // 8-byte chunks per pixel
    constexpr int TOT = 4 * 66 * CHUNKS;
    const size_t planeBase = (size_t)b * H * (size_t)W;
    for (int i = tid; i < TOT; i += blockDim.x) {
      int chunk = i % CHUNKS;
      int pxl   = i / CHUNKS;
      int col = pxl % 66;
      int row = pxl / 66;
      int gy = y0 + row - 1;
      int gx = x0 + col - 1;
      uint2 v = make_uint2(0u, 0u);
      if (gy >= 0 && gy < H && gx >= 0 && gx < W) {
        const unsigned short* p = in + (planeBase + (size_t)gy * W + gx) * CP + chunk * 4;
        v = *(const uint2*)p;
      }
      *(uint2*)(ldsIn + (size_t)i * 8) = v;
    }
  }
  // Stage packed weights.
  for (int i = tid * 16; i < W_BYTES; i += blockDim.x * 16)
    *(uint4*)(ldsW + i) = *(const uint4*)((const char*)wpk + i);
  __syncthreads();

  const int lane = tid & 31;
  const int wave = tid >> 5;
  const int mt   = wave & 7;
  const int rsub = mt >> 2;         // which of the 2 output rows
  const int xt   = mt & 3;          // 16-px x-tile
  const int ng   = wave >> 3;       // N-group
  const int nt0  = 2 * ng;
  const int nt1  = 2 * ng + 1;
  const int hi = (lane >> 4) & 1;
  const int lm = lane & 15;

  const char* aPixBase = ldsIn + (size_t)(xt * 16 + lm) * (CP * 2) + hi * 16;
  const char* bBase    = ldsW + (size_t)lm * 64 + hi * 16;

  auto aPtr = [&](int s) -> const char* {
    int tap = s / KC, kc = s % KC;
    int dy = tap / 3, dx = tap % 3;
    return aPixBase + (size_t)((rsub + dy) * 66 + dx) * (CP * 2) + kc * 64;
  };
  auto bPtr = [&](int s, int nt) -> const char* {
    return bBase + (size_t)(s * OP + nt * 16) * 64;
  };

  FragU fa[2], f0[2], f1[2];
  auto loadFrags = [&](int s, int buf) {
    const char* ap = aPtr(s);
    fa[buf].u[0] = *(const uint4*)(ap);
    fa[buf].u[1] = *(const uint4*)(ap + 32);
    const char* bp0 = bPtr(s, nt0);
    f0[buf].u[0] = *(const uint4*)(bp0);
    f0[buf].u[1] = *(const uint4*)(bp0 + 32);
    if (DUAL) {
      const char* bp1 = bPtr(s, nt1);
      f1[buf].u[0] = *(const uint4*)(bp1);
      f1[buf].u[1] = *(const uint4*)(bp1 + 32);
    }
  };

  v8f_t acc0 = {0.f,0.f,0.f,0.f,0.f,0.f,0.f,0.f};
  v8f_t acc1 = {0.f,0.f,0.f,0.f,0.f,0.f,0.f,0.f};
  loadFrags(0, 0);
#pragma unroll
  for (int s = 0; s < STEPS; ++s) {
    const int cur = s & 1;
    if (s + 1 < STEPS) loadFrags(s + 1, cur ^ 1);   // prefetch into other regs
    acc0 = WMMA_BF16(fa[cur].v, f0[cur].v, acc0);
    if (DUAL)
      acc1 = WMMA_BF16(fa[cur].v, f1[cur].v, acc1); // independent acc chain
  }

  const int y = y0 + rsub;
  const size_t rowBase = ((size_t)b * H + y) * (size_t)W;
  auto storeTile = [&](v8f_t& acc, int nt) {
    const int n = nt * 16 + lm;
    const float bn = (n < bias_n) ? bias[n] : 0.f;
#pragma unroll
    for (int r = 0; r < 8; ++r) {
      int px = x0 + xt * 16 + r + hi * 8;           // C layout: M = r (+8 hi lanes)
      float v = acc[r] + bn;
      size_t o = (rowBase + px) * OP + n;
      if (resid) v += bf2f(resid[o]);
      if (relu)  v = fmaxf(v, 0.f);
      out[o] = f2bf(v);
    }
  };
  storeTile(acc0, nt0);
  if (DUAL) storeTile(acc1, nt1);
}

// ---------------------------------------------------------------------------
// 1x1 conv as GEMM over pixels: [M, KC*32] x [KC*32, NT*16]. Block: 128 px.
// A fragments loaded once per wave; N-tiles processed in independent pairs.
// EPI: 0 none, 1 relu, 2 exact gelu, 3 multiply by 2x-gathered feat.
// ---------------------------------------------------------------------------
template<int KC, int NT, int EPI>
__global__ void __launch_bounds__(256)
gemm1x1_wmma(const unsigned short* __restrict__ in,   // [M,KC*32] bf16
             const unsigned short* __restrict__ wpk,  // [KC][NT*16][32] bf16
             const float* __restrict__ bias, int bias_n,
             unsigned short* __restrict__ out,        // [M,NT*16] bf16
             const unsigned short* __restrict__ feat) // [B,192,192,64] for EPI==3
{
  constexpr int KP = KC * 32;
  constexpr int NP = NT * 16;
  constexpr int A_BYTES = 128 * KP * 2;
  constexpr int W_BYTES = KC * NP * 64;
  __shared__ char smem[A_BYTES + W_BYTES];
  const size_t m0 = (size_t)blockIdx.x * 128;
  char* ldsA = smem;
  char* ldsW = smem + A_BYTES;

  {
    const char* src = (const char*)(in + m0 * KP);
    for (int i = threadIdx.x * 16; i < A_BYTES; i += blockDim.x * 16)
      *(uint4*)(ldsA + i) = *(const uint4*)(src + i);
    for (int i = threadIdx.x * 16; i < W_BYTES; i += blockDim.x * 16)
      *(uint4*)(ldsW + i) = *(const uint4*)((const char*)wpk + i);
  }
  __syncthreads();

  const int lane = threadIdx.x & 31;
  const int wave = threadIdx.x >> 5;     // m-tile 0..7
  const int hi = (lane >> 4) & 1;
  const int lm = lane & 15;

  FragU fa[KC];
  {
    const char* aBase = ldsA + (size_t)(wave * 16 + lm) * (KP * 2) + hi * 16;
#pragma unroll
    for (int kc = 0; kc < KC; ++kc) {
      fa[kc].u[0] = *(const uint4*)(aBase + kc * 64);
      fa[kc].u[1] = *(const uint4*)(aBase + kc * 64 + 32);
    }
  }
  const char* bBase = ldsW + (size_t)lm * 64 + hi * 16;

  auto storeTile = [&](v8f_t& acc, int nt) {
    const int n = nt * 16 + lm;
    const float bn = (n < bias_n) ? bias[n] : 0.f;
#pragma unroll
    for (int r = 0; r < 8; ++r) {
      size_t px = m0 + (size_t)(wave * 16 + r + hi * 8);
      float v = acc[r] + bn;
      if (EPI == 1)      v = fmaxf(v, 0.f);
      else if (EPI == 2) v = 0.5f * v * (1.f + erff(v * 0.70710678118654752f));
      else if (EPI == 3) {
        size_t bb  = px / (384u * 384u);
        size_t rem = px - bb * (384u * 384u);
        int yy = (int)(rem / 384u), xx = (int)(rem % 384u);
        v *= bf2f(feat[((bb * 192 + (yy >> 1)) * 192 + (xx >> 1)) * 64 + n]);
      }
      out[px * NP + n] = f2bf(v);
    }
  };

#pragma unroll
  for (int ntp = 0; ntp < NT / 2; ++ntp) {
    v8f_t acc0 = {0.f,0.f,0.f,0.f,0.f,0.f,0.f,0.f};
    v8f_t acc1 = {0.f,0.f,0.f,0.f,0.f,0.f,0.f,0.f};
#pragma unroll
    for (int kc = 0; kc < KC; ++kc) {
      FragU b0, b1;
      const char* bp0 = bBase + (size_t)(kc * NP + (2 * ntp) * 16) * 64;
      const char* bp1 = bBase + (size_t)(kc * NP + (2 * ntp + 1) * 16) * 64;
      b0.u[0] = *(const uint4*)(bp0);
      b0.u[1] = *(const uint4*)(bp0 + 32);
      b1.u[0] = *(const uint4*)(bp1);
      b1.u[1] = *(const uint4*)(bp1 + 32);
      acc0 = WMMA_BF16(fa[kc].v, b0.v, acc0);
      acc1 = WMMA_BF16(fa[kc].v, b1.v, acc1);
    }
    storeTile(acc0, 2 * ntp);
    storeTile(acc1, 2 * ntp + 1);
  }
}

// ---------------------------------------------------------------------------
// Weight packers (fp32 -> bf16 WMMA-B tile layout, zero padded).
// ---------------------------------------------------------------------------
__global__ void pack_conv3x3(const float* __restrict__ src, unsigned short* __restrict__ dst,
                             int Creal, int Oreal, int KC, int Opad)
{
  int total = 9 * KC * Opad * 32;
  for (int i = blockIdx.x * blockDim.x + threadIdx.x; i < total; i += gridDim.x * blockDim.x) {
    int ki = i & 31;
    int t = i >> 5;
    int o = t % Opad; t /= Opad;
    int kc = t % KC;
    int tap = t / KC;
    int c = kc * 32 + ki;
    float v = (c < Creal && o < Oreal) ? src[(o * Creal + c) * 9 + tap] : 0.f;
    dst[i] = f2bf(v);
  }
}

__global__ void pack_conv1x1(const float* __restrict__ src, unsigned short* __restrict__ dst,
                             int Creal, int Nreal, int KC, int Npad)
{
  int total = KC * Npad * 32;
  for (int i = blockIdx.x * blockDim.x + threadIdx.x; i < total; i += gridDim.x * blockDim.x) {
    int ki = i & 31;
    int t = i >> 5;
    int n = t % Npad;
    int kc = t / Npad;
    int c = kc * 32 + ki;
    float v = (c < Creal && n < Nreal) ? src[n * Creal + c] : 0.f;
    dst[i] = f2bf(v);
  }
}

// ---------------------------------------------------------------------------
// Layout conversion / gather kernels.
// ---------------------------------------------------------------------------
__global__ void in_to_nhwc(const float* __restrict__ x, unsigned short* __restrict__ X0)
{
  size_t idx = (size_t)blockIdx.x * blockDim.x + threadIdx.x;
  const size_t total = (size_t)8 * 192 * 192;
  if (idx >= total) return;
  size_t b = idx / (192u * 192u);
  size_t rem = idx - b * (192u * 192u);
  int yy = (int)(rem / 192u), xx = (int)(rem % 192u);
  union { unsigned short s[32]; uint4 u[4]; } o;
#pragma unroll
  for (int c = 0; c < 32; ++c) o.s[c] = 0;
  for (int c = 0; c < 3; ++c)
    o.s[c] = f2bf(x[((b * 3 + c) * 192 + yy) * 192 + xx]);
  uint4* dst = (uint4*)(X0 + idx * 32);
#pragma unroll
  for (int i = 0; i < 4; ++i) dst[i] = o.u[i];
}

// cat[:, :24] = x_red gathered 2x-nearest; cat[:, 24:48] = relu(ext(frac)); rest 0.
__global__ void gather_cat(const unsigned short* __restrict__ xred,  // [B*192*192,32]
                           const float* __restrict__ extw,           // [24,2]
                           const float* __restrict__ extb,           // [24]
                           unsigned short* __restrict__ cat)         // [B*384*384,64]
{
  size_t idx = (size_t)blockIdx.x * blockDim.x + threadIdx.x;
  const size_t total = (size_t)8 * 384 * 384;
  if (idx >= total) return;
  size_t b = idx / (384u * 384u);
  size_t rem = idx - b * (384u * 384u);
  int yy = (int)(rem / 384u), xx = (int)(rem % 384u);
  const unsigned short* src = xred + ((b * 192 + (yy >> 1)) * 192 + (xx >> 1)) * 32;
  union { unsigned short s[64]; uint4 u[8]; } o;
#pragma unroll
  for (int c = 0; c < 24; ++c) o.s[c] = src[c];
  float fy = 0.5f * (yy & 1), fx = 0.5f * (xx & 1);
  for (int c = 0; c < 24; ++c) {
    float v = extb[c] + extw[2 * c] * fy + extw[2 * c + 1] * fx;
    o.s[24 + c] = f2bf(fmaxf(v, 0.f));
  }
#pragma unroll
  for (int c = 48; c < 64; ++c) o.s[c] = 0;
  uint4* dst = (uint4*)(cat + idx * 64);
#pragma unroll
  for (int i = 0; i < 8; ++i) dst[i] = o.u[i];
}

__global__ void tail_to_out(const unsigned short* __restrict__ tail,  // [B*384*384,16]
                            float* __restrict__ out)                  // [8,3,384,384]
{
  size_t idx = (size_t)blockIdx.x * blockDim.x + threadIdx.x;
  const size_t total = (size_t)8 * 384 * 384;
  if (idx >= total) return;
  size_t b = idx / (384u * 384u);
  size_t rem = idx - b * (384u * 384u);
  int yy = (int)(rem / 384u), xx = (int)(rem % 384u);
  for (int c = 0; c < 3; ++c)
    out[((b * 3 + c) * 384 + yy) * 384 + xx] = bf2f(tail[idx * 16 + c]);
}

// ---------------------------------------------------------------------------
// Host orchestration.
// ---------------------------------------------------------------------------
extern "C" void kernel_launch(void* const* d_in, const int* in_sizes, int n_in,
                              void* d_out, int out_size, void* d_ws, size_t ws_size,
                              hipStream_t stream)
{
  (void)in_sizes; (void)n_in; (void)out_size; (void)ws_size;
  // d_in: [0]=x, [1]=out_h, [2]=out_w, then params in make_params() order.
  const float* x_in = (const float*)d_in[0];
  char* ws = (char*)d_ws;

  size_t off = 0;
  auto alloc = [&](size_t bytes) { size_t o = off; off += (bytes + 255) & ~(size_t)255; return o; };

  // Weight slots: [0]=enc_head, [1..8]=enc b{i}{w1,w2}, [9]=dec_head,
  //               [10..17]=dec b{i}{w1,w2}, [18]=tail.
  size_t wslot[19];
  wslot[0] = alloc((size_t)9 * 1 * 64 * 32 * 2);
  for (int i = 1; i <= 17; ++i) wslot[i] = alloc((size_t)9 * 2 * 64 * 32 * 2);
  wslot[18] = alloc((size_t)9 * 2 * 16 * 32 * 2);
  size_t w_red = alloc((size_t)2 * 32 * 32 * 2);
  size_t w_p1  = alloc((size_t)2 * 96 * 32 * 2);
  size_t w_p2  = alloc((size_t)3 * 64 * 32 * 2);

  const size_t PX192 = (size_t)8 * 192 * 192;
  const size_t PX384 = (size_t)8 * 384 * 384;
  size_t X0  = alloc(PX192 * 32 * 2);
  size_t FA  = alloc(PX192 * 64 * 2);
  size_t FB  = alloc(PX192 * 64 * 2);
  size_t R1  = alloc(PX192 * 64 * 2);
  size_t XR  = alloc(PX192 * 32 * 2);
  size_t CAT = alloc(PX384 * 64 * 2);
  size_t H96 = alloc(PX384 * 96 * 2);
  size_t PRJ = alloc(PX384 * 64 * 2);
  // Reuse dead regions for the decoder stage (peak ~680 MB workspace).
  size_t DA = CAT, DB = H96, R3 = PRJ, TAIL = R1;

  auto wsu16 = [&](size_t o) { return (unsigned short*)(ws + o); };

  // ---- pack weights ----
  auto pack3 = [&](int din, size_t dst, int C, int O, int KC, int Opad) {
    int total = 9 * KC * Opad * 32;
    pack_conv3x3<<<dim3((total + 255) / 256), dim3(256), 0, stream>>>(
        (const float*)d_in[din], wsu16(dst), C, O, KC, Opad);
  };
  pack3(3, wslot[0], 3, 48, 1, 64);                        // enc_head
  for (int i = 0; i < 4; ++i) {
    pack3(5 + 4 * i,     wslot[1 + 2 * i], 48, 48, 2, 64); // enc b{i} w1
    pack3(5 + 4 * i + 2, wslot[2 + 2 * i], 48, 48, 2, 64); // enc b{i} w2
  }
  pack3(21, wslot[9], 48, 48, 2, 64);                      // dec_head
  for (int i = 0; i < 4; ++i) {
    pack3(23 + 4 * i,     wslot[10 + 2 * i], 48, 48, 2, 64);
    pack3(23 + 4 * i + 2, wslot[11 + 2 * i], 48, 48, 2, 64);
  }
  pack3(39, wslot[18], 48, 3, 2, 16);                      // tail
  auto pack1 = [&](int din, size_t dst, int C, int N, int KC, int Npad) {
    int total = KC * Npad * 32;
    pack_conv1x1<<<dim3((total + 255) / 256), dim3(256), 0, stream>>>(
        (const float*)d_in[din], wsu16(dst), C, N, KC, Npad);
  };
  pack1(41, w_red, 48, 24, 2, 32);
  pack1(45, w_p1,  48, 96, 2, 96);
  pack1(47, w_p2,  96, 48, 3, 64);

  // ---- conv helpers (templated dispatch) ----
  auto conv48 = [&](size_t inO, size_t wO, int biasIdx, size_t residO, bool useRes,
                    size_t outO, int H, int W, int relu) {
    dim3 grid(W / 64, H / 2, 8);
    conv3x3_wmma<2, 4><<<grid, dim3(512), 0, stream>>>(
        wsu16(inO), wsu16(wO), (const float*)d_in[biasIdx], 48,
        useRes ? wsu16(residO) : nullptr, wsu16(outO), H, W, relu);
  };

  // ---- encoder @192 ----
  in_to_nhwc<<<dim3((unsigned)((PX192 + 255) / 256)), dim3(256), 0, stream>>>(x_in, wsu16(X0));
  conv3x3_wmma<1, 4><<<dim3(3, 96, 8), dim3(512), 0, stream>>>(
      wsu16(X0), wsu16(wslot[0]), (const float*)d_in[4], 48,
      nullptr, wsu16(FA), 192, 192, 0);
  size_t cur = FA, alt = FB;
  for (int i = 0; i < 4; ++i) {
    conv48(cur, wslot[1 + 2 * i], 5 + 4 * i + 1, 0, false, R1, 192, 192, 1);
    conv48(R1, wslot[2 + 2 * i], 5 + 4 * i + 3, cur, true, alt, 192, 192, 0);
    size_t t = cur; cur = alt; alt = t;
  }
  // cur == FA holds the 48-ch encoder features.

  // ---- modulation path ----
  gemm1x1_wmma<2, 2, 1><<<dim3((unsigned)(PX192 / 128)), dim3(256), 0, stream>>>(
      wsu16(cur), wsu16(w_red), (const float*)d_in[42], 24, wsu16(XR), nullptr);
  gather_cat<<<dim3((unsigned)((PX384 + 255) / 256)), dim3(256), 0, stream>>>(
      wsu16(XR), (const float*)d_in[43], (const float*)d_in[44], wsu16(CAT));
  gemm1x1_wmma<2, 6, 2><<<dim3((unsigned)(PX384 / 128)), dim3(256), 0, stream>>>(
      wsu16(CAT), wsu16(w_p1), (const float*)d_in[46], 96, wsu16(H96), nullptr);
  gemm1x1_wmma<3, 4, 3><<<dim3((unsigned)(PX384 / 128)), dim3(256), 0, stream>>>(
      wsu16(H96), wsu16(w_p2), (const float*)d_in[48], 48, wsu16(PRJ), wsu16(cur));

  // ---- decoder @384 ----
  conv48(PRJ, wslot[9], 22, 0, false, DA, 384, 384, 0);
  size_t dcur = DA, dalt = DB;
  for (int i = 0; i < 4; ++i) {
    conv48(dcur, wslot[10 + 2 * i], 23 + 4 * i + 1, 0, false, R3, 384, 384, 1);
    conv48(R3, wslot[11 + 2 * i], 23 + 4 * i + 3, dcur, true, dalt, 384, 384, 0);
    size_t t = dcur; dcur = dalt; dalt = t;
  }
  conv3x3_wmma<2, 1><<<dim3(6, 192, 8), dim3(256), 0, stream>>>(
      wsu16(dcur), wsu16(wslot[18]), (const float*)d_in[40], 3,
      nullptr, wsu16(TAIL), 384, 384, 0);

  tail_to_out<<<dim3((unsigned)((PX384 + 255) / 256)), dim3(256), 0, stream>>>(
      wsu16(TAIL), (float*)d_out);
}
